// MSFN_47425028883133
// MI455X (gfx1250) — matrix operand order repaired and checked by reading, not compiled, not run on hardware
//
#include <hip/hip_runtime.h>
#include <hip/hip_bf16.h>

typedef __bf16 v16bf __attribute__((ext_vector_type(16)));
typedef __bf16 v8bf  __attribute__((ext_vector_type(8)));
typedef float  v8f   __attribute__((ext_vector_type(8)));

// ---- WMMA 16x16x32 bf16 lane layouts (CDNA5 ISA 7.12.2) ----
// A (16xK): lanes 0-15 & 16-31 hold M=0..15; lane's 16 elems = K runs
//   [kb + hi*16 .. +7] and [kb + hi*16 + 8 .. +7]  (two contiguous 16B runs)
// B (Kx16): lane15 = N; lane's 16 elems = K run [kb + hi*16 .. +15] (one 32B run)
// D: VGPR r -> M = mBase + r + 8*hi, N = nBase + l15
static __device__ __forceinline__ int akidx(int v, int hi) {
    return 2 * v + (hi ? 16 : 0);
}
static __device__ __forceinline__ int bkidx(int v, int hi) {
    return 2 * v + (hi ? 16 : 0);
}

// D = A(MxK, row-major, 16B-aligned rows) * Bp(packed fragments), one 16x16 tile.
static __device__ __forceinline__ v8f wmma_tile_packed(const __bf16* __restrict__ A, int lda,
                                                       const __bf16* __restrict__ Bp,
                                                       int mBase, int ni, int K)
{
    const int lane = threadIdx.x & 31;
    const int hi   = lane >> 4;
    const int l15  = lane & 15;
    const __bf16* Arow = A + (size_t)(mBase + l15) * lda + (hi ? 16 : 0);
    const __bf16* Bl   = Bp + ((size_t)ni * (K >> 5) * 32 + lane) * 16;
    v8f acc = {0.f, 0.f, 0.f, 0.f, 0.f, 0.f, 0.f, 0.f};
    for (int kb = 0; kb < K; kb += 32) {
        v8bf  a0 = *(const v8bf*)(Arow + kb);
        v8bf  a1 = *(const v8bf*)(Arow + kb + 8);
        v16bf bb = *(const v16bf*)Bl;
        Bl += 32 * 16;
        __builtin_prefetch(Bl, 0, 0);
        v16bf a;
#pragma unroll
        for (int j = 0; j < 8; ++j) { a[j] = a0[j]; a[8 + j] = a1[j]; }
        acc = __builtin_amdgcn_wmma_f32_16x16x32_bf16(false, a, false, bb,
                                                      (short)0, acc, false, false);
    }
    return acc;
}

static __device__ __forceinline__ void store_tile_bf16(__bf16* __restrict__ D, int ldd,
                                                       int mBase, int nBase, v8f acc,
                                                       int lrelu)
{
    const int lane = threadIdx.x & 31;
    const int hi   = lane >> 4;
    const int l15  = lane & 15;
#pragma unroll
    for (int r = 0; r < 8; ++r) {
        float v = acc[r];
        if (lrelu) v = (v >= 0.f) ? v : 0.01f * v;
        D[(size_t)(mBase + r + 8 * hi) * ldd + nBase + l15] = (__bf16)v;
    }
}

// ---- f32 -> bf16 row-major convert (A operands) ----
__global__ void cvt512_kernel(const float* __restrict__ src, __bf16* __restrict__ dst)
{
    int i = blockIdx.x * blockDim.x + threadIdx.x;
    if (i >= 512 * 512) return;
    dst[i] = (__bf16)src[i];
}

// ---- pack B (KxN) fp32 into WMMA-B fragment order (coalesced writes) ----
// dst[((ni*(K/32)+kb)*32+lane)*16+e] = B[kb*32 + e + ((lane&16)?16:0)][ni*16 + (lane&15)]
// transpose=0: B[k][n]=src[k*N+n]; transpose=1: B[k][n]=src[n*K+k]
__global__ void pack_b_f32_kernel(const float* __restrict__ src, __bf16* __restrict__ dst,
                                  int K, int N, int transpose)
{
    int i = blockIdx.x * blockDim.x + threadIdx.x;
    if (i >= K * N) return;
    int e    = i & 15;
    int lane = (i >> 4) & 31;
    int kb   = (i >> 9) % (K >> 5);
    int ni   = (i >> 9) / (K >> 5);
    int k = kb * 32 + e + ((lane & 16) ? 16 : 0);
    int n = ni * 16 + (lane & 15);
    float v = transpose ? src[(size_t)n * K + k] : src[(size_t)k * N + n];
    dst[i] = (__bf16)v;
}

// ---- pack X_b (512x384) directly from out1..3 into WMMA-B fragments ----
// X_b[k=c][n=t*3+s] = out_s[b][c][t]
__global__ void pack_x_kernel(const float* __restrict__ o1, const float* __restrict__ o2,
                              const float* __restrict__ o3, __bf16* __restrict__ Xp)
{
    const int b = blockIdx.y;
    int i = blockIdx.x * blockDim.x + threadIdx.x; // 512*384
    if (i >= 512 * 384) return;
    int e    = i & 15;
    int lane = (i >> 4) & 31;
    int kb   = (i >> 9) & 15;  // K/32 = 16
    int ni   = (i >> 9) >> 4;  // 0..23
    int c = kb * 32 + e + ((lane & 16) ? 16 : 0);
    int n = ni * 16 + (lane & 15);
    int t = n / 3, s = n - 3 * t;
    const float* xs = (s == 0) ? o1 : ((s == 1) ? o2 : o3);
    Xp[(size_t)b * 512 * 384 + i] = (__bf16)xs[((size_t)b * 512 + c) * 128 + t];
}

// ---- generic WMMA GEMM (packed B), one wave per 16x16 tile ----
__global__ void gemm_bf16_kernel(const __bf16* __restrict__ A, const __bf16* __restrict__ Bp,
                                 __bf16* __restrict__ D, int M, int N, int K, int lrelu)
{
    int wave   = (blockIdx.x * blockDim.x + threadIdx.x) >> 5;
    int ntiles = N >> 4;
    int mi = wave / ntiles, ni = wave % ntiles;
    if (mi >= (M >> 4)) return;
    v8f acc = wmma_tile_packed(A, K, Bp, mi * 16, ni, K);
    store_tile_bf16(D, N, mi * 16, ni * 16, acc, lrelu);
}

// ---- projections: theta/phi stored TRANSPOSED [col][o] (v8bf store), wg row-major ----
__global__ void proj_kernel(const __bf16* __restrict__ thw, const __bf16* __restrict__ phw,
                            const __bf16* __restrict__ wgw, const __bf16* __restrict__ Xp,
                            __bf16* __restrict__ P)
{
    const int z = blockIdx.y;       // 0..47 : p*16 + b
    const int p = z >> 4, b = z & 15;
    const __bf16* W  = (p == 0) ? thw : ((p == 1) ? phw : wgw);
    const __bf16* Bp = Xp + (size_t)b * 512 * 384;
    __bf16*       D  = P + (size_t)z * 512 * 384;
    int wave = (blockIdx.x * blockDim.x + threadIdx.x) >> 5;
    int mi = wave / 24, ni = wave % 24; // 32 x 24 tiles
    if (mi >= 32) return;
    v8f acc = wmma_tile_packed(W, 512, Bp, mi * 16, ni, 512);
    const int lane = threadIdx.x & 31;
    const int hi   = lane >> 4;
    const int l15  = lane & 15;
    if (p < 2) {
        v8bf row;
#pragma unroll
        for (int r = 0; r < 8; ++r) row[r] = (__bf16)acc[r];
        *(v8bf*)(D + (size_t)(ni * 16 + l15) * 512 + mi * 16 + 8 * hi) = row;
    } else {
        store_tile_bf16(D, 384, mi * 16, ni * 16, acc, 0);
    }
}

// ---- per-(b,t) window attention ----
__global__ void window_attn_kernel(const __bf16* __restrict__ P,
                                   const float* __restrict__ x0,
                                   const float* __restrict__ x1,
                                   const float* __restrict__ x2,
                                   const float* __restrict__ Wb,
                                   __bf16* __restrict__ pooled)
{
    const int bid = blockIdx.x;      // 0..2047
    const int b   = bid >> 7;
    const int t   = bid & 127;
    const int colbase = 3 * (t - 7); // contiguous 24-col window, left-clamped

    const size_t PS = (size_t)512 * 384;
    const __bf16* PthT = P + (size_t)(b) * PS;       // [col][o], ld 512
    const __bf16* PphT = P + (size_t)(16 + b) * PS;  // [col][o], ld 512
    const __bf16* Pwg  = P + (size_t)(32 + b) * PS;  // [o][col], ld 384

    __shared__ float  fbuf[32 * 32];
    __shared__ __bf16 fT[32 * 32];   // fT[m][n] = softmax(f)[n][m], zero padded

    const int tid  = threadIdx.x;    // 128 threads = 4 waves
    const int wave = tid >> 5;
    const int lane = tid & 31;
    const int hi   = lane >> 4;
    const int l15  = lane & 15;
    const __bf16 zb = (__bf16)0.0f;

    for (int i = tid; i < 1024; i += 128) fT[i] = zb;

    // phase 1: f[n,m] = sum_o theta_win[o,n] * phi_win[o,m]  (pad 24 -> 32)
    // transposed P => A and B operands are contiguous vector loads.
    {
        const int mi = wave >> 1;    // n tile
        const int ni = wave & 1;     // m tile
        const int n = mi * 16 + l15;
        const int m = ni * 16 + l15;
        const bool nval = (n < 24) && (colbase + n >= 0);
        const bool mval = (m < 24) && (colbase + m >= 0);
        const int arow = nval ? (colbase + n) : 0;   // clamped: loads stay in-bounds
        const int brow = mval ? (colbase + m) : 0;
        const __bf16* Ap = PthT + (size_t)arow * 512 + (hi ? 16 : 0);
        const __bf16* Bq = PphT + (size_t)brow * 512 + (hi ? 16 : 0);
        v8f acc = {0.f, 0.f, 0.f, 0.f, 0.f, 0.f, 0.f, 0.f};
        for (int kb = 0; kb < 512; kb += 32) {
            v8bf  a0 = *(const v8bf*)(Ap + kb);
            v8bf  a1 = *(const v8bf*)(Ap + kb + 8);
            v16bf bl = *(const v16bf*)(Bq + kb);
            v16bf a, bb;
#pragma unroll
            for (int j = 0; j < 8; ++j) {
                a[j]     = nval ? a0[j] : zb;
                a[8 + j] = nval ? a1[j] : zb;
            }
#pragma unroll
            for (int j = 0; j < 16; ++j) bb[j] = mval ? bl[j] : zb;
            acc = __builtin_amdgcn_wmma_f32_16x16x32_bf16(false, a, false, bb,
                                                          (short)0, acc, false, false);
        }
#pragma unroll
        for (int r = 0; r < 8; ++r)
            fbuf[(mi * 16 + r + 8 * hi) * 32 + ni * 16 + l15] = acc[r];
    }
    __syncthreads();

    // softmax over m for rows n < 24; store transposed bf16
    if (tid < 24) {
        const int n = tid;
        float mx = -1e30f;
        for (int m = 0; m < 24; ++m) mx = fmaxf(mx, fbuf[n * 32 + m]);
        float s = 0.f;
        for (int m = 0; m < 24; ++m) s += __expf(fbuf[n * 32 + m] - mx);
        const float inv = 1.0f / s;
        for (int m = 0; m < 24; ++m)
            fT[m * 32 + n] = (__bf16)(__expf(fbuf[n * 32 + m] - mx) * inv);
    }
    __syncthreads();

    // phase 2: Wy[c,n] = sum_m WG_win[c,m] * fT[m][n]; z = Wy + Wb + X; max over n
    for (int mt = wave; mt < 32; mt += 4) {
        const int crow = mt * 16 + l15;  // A-row channel
        float vmax[8];
#pragma unroll
        for (int r = 0; r < 8; ++r) vmax[r] = -1e30f;

#pragma unroll
        for (int nt = 0; nt < 2; ++nt) {
            v16bf a, bb;
#pragma unroll
            for (int v = 0; v < 8; ++v) {
                const int ka = akidx(v, hi);                 // k = m (0..31)
                const bool v0 = (ka     < 24) && (colbase + ka     >= 0);
                const bool v1 = (ka + 1 < 24) && (colbase + ka + 1 >= 0);
                a[2*v]   = v0 ? Pwg[(size_t)crow * 384 + colbase + ka]     : zb;
                a[2*v+1] = v1 ? Pwg[(size_t)crow * 384 + colbase + ka + 1] : zb;
                const int kv = bkidx(v, hi);
                bb[2*v]   = fT[kv * 32 + nt * 16 + l15];
                bb[2*v+1] = fT[(kv + 1) * 32 + nt * 16 + l15];
            }
            v8f acc = {0.f, 0.f, 0.f, 0.f, 0.f, 0.f, 0.f, 0.f};
            acc = __builtin_amdgcn_wmma_f32_16x16x32_bf16(false, a, false, bb,
                                                          (short)0, acc, false, false);
            const int n = nt * 16 + l15;
            if (n < 24) {
                const int j  = n / 3;
                const int s  = n - 3 * j;
                const int tt = t - 7 + j;
                const float* xs = (s == 0) ? x0 : ((s == 1) ? x1 : x2);
#pragma unroll
                for (int r = 0; r < 8; ++r) {
                    const int c = mt * 16 + r + 8 * hi;
                    const float xv = (tt >= 0) ? xs[((size_t)b * 512 + c) * 128 + tt] : 0.0f;
                    vmax[r] = fmaxf(vmax[r], acc[r] + xv);
                }
            }
        }
#pragma unroll
        for (int r = 0; r < 8; ++r) {
            float v = vmax[r];
#pragma unroll
            for (int off = 1; off < 16; off <<= 1)
                v = fmaxf(v, __shfl_xor(v, off, 32));
            vmax[r] = v;
        }
        if (l15 == 0) {
#pragma unroll
            for (int r = 0; r < 8; ++r) {
                const int c = mt * 16 + r + 8 * hi;
                pooled[(size_t)(t * 16 + b) * 512 + c] = (__bf16)(Wb[c] + vmax[r]);
            }
        }
    }
}

// ---- final 1-wide projection + lrelu, fp32 out (T,B,1) ----
__global__ void head3_kernel(const __bf16* __restrict__ H2, const float* __restrict__ p3w,
                             const float* __restrict__ p3b, float* __restrict__ out)
{
    int wave = (blockIdx.x * blockDim.x + threadIdx.x) >> 5;
    int lane = threadIdx.x & 31;
    if (wave >= 2048) return;
    const __bf16* row = H2 + (size_t)wave * 512;
    float s = 0.f;
    for (int c = lane; c < 512; c += 32) s += (float)row[c] * p3w[c];
#pragma unroll
    for (int off = 16; off > 0; off >>= 1) s += __shfl_xor(s, off, 32);
    if (lane == 0) {
        float v = s + p3b[0];
        out[wave] = (v >= 0.f) ? v : 0.01f * v;
    }
}

extern "C" void kernel_launch(void* const* d_in, const int* in_sizes, int n_in,
                              void* d_out, int out_size, void* d_ws, size_t ws_size,
                              hipStream_t stream)
{
    const float* out1    = (const float*)d_in[0];
    const float* out2    = (const float*)d_in[1];
    const float* out3    = (const float*)d_in[2];
    const float* theta_w = (const float*)d_in[3];
    const float* phi_w   = (const float*)d_in[4];
    const float* g_w     = (const float*)d_in[5];
    const float* W_w     = (const float*)d_in[6];
    const float* W_b     = (const float*)d_in[7];
    const float* p1_w    = (const float*)d_in[8];
    const float* p2_w    = (const float*)d_in[9];
    const float* p3_w    = (const float*)d_in[10];
    const float* p3_b    = (const float*)d_in[11];
    float* outp = (float*)d_out;

    char* ws = (char*)d_ws;
    size_t off = 0;
    auto alloc = [&](size_t bytes) { char* p = ws + off; off += (bytes + 255) & ~(size_t)255; return p; };
    __bf16* Xp     = (__bf16*)alloc((size_t)16 * 512 * 384 * 2); // packed X fragments
    __bf16* th_bf  = (__bf16*)alloc(512 * 512 * 2);              // A row-major
    __bf16* ph_bf  = (__bf16*)alloc(512 * 512 * 2);
    __bf16* Ww_bf  = (__bf16*)alloc(512 * 512 * 2);
    __bf16* gwP    = (__bf16*)alloc(512 * 512 * 2);              // g_w packed as B
    __bf16* Wg_bf  = (__bf16*)alloc(512 * 512 * 2);              // W_w @ g_w (row-major)
    __bf16* p1P    = (__bf16*)alloc(512 * 512 * 2);              // p1_w^T packed as B
    __bf16* p2P    = (__bf16*)alloc(512 * 512 * 2);              // p2_w^T packed as B
    __bf16* P      = (__bf16*)alloc((size_t)48 * 512 * 384 * 2); // thetaT/phiT/wg projections
    __bf16* pooled = (__bf16*)alloc((size_t)2048 * 512 * 2);
    __bf16* H1     = (__bf16*)alloc((size_t)2048 * 512 * 2);
    __bf16* H2     = (__bf16*)alloc((size_t)2048 * 512 * 2);
    (void)ws_size; (void)in_sizes; (void)n_in; (void)out_size;

    // stage 0: converts + packs
    cvt512_kernel<<<1024, 256, 0, stream>>>(theta_w, th_bf);
    cvt512_kernel<<<1024, 256, 0, stream>>>(phi_w,   ph_bf);
    cvt512_kernel<<<1024, 256, 0, stream>>>(W_w,     Ww_bf);
    pack_b_f32_kernel<<<1024, 256, 0, stream>>>(g_w,  gwP, 512, 512, 0); // B[k][c]=g_w[k][c]
    pack_b_f32_kernel<<<1024, 256, 0, stream>>>(p1_w, p1P, 512, 512, 1); // B[c][o]=p1_w[o][c]
    pack_b_f32_kernel<<<1024, 256, 0, stream>>>(p2_w, p2P, 512, 512, 1);
    pack_x_kernel<<<dim3(768, 16), 256, 0, stream>>>(out1, out2, out3, Xp);

    // stage 1: fuse W into g: Wg = W_w @ g_w
    gemm_bf16_kernel<<<128, 256, 0, stream>>>(Ww_bf, gwP, Wg_bf, 512, 512, 512, 0);

    // stage 2: projections P[p][b] = W_p @ X_b  (theta/phi transposed, wg row-major)
    proj_kernel<<<dim3(96, 48), 256, 0, stream>>>(th_bf, ph_bf, Wg_bf, Xp, P);

    // stage 3: per-window attention + residual + max-pool
    window_attn_kernel<<<2048, 128, 0, stream>>>(P, out1, out2, out3, W_b, pooled);

    // stage 4: MLP head with fused leaky-relu
    gemm_bf16_kernel<<<512, 256, 0, stream>>>(pooled, p1P, H1, 2048, 512, 512, 1);
    gemm_bf16_kernel<<<512, 256, 0, stream>>>(H1,     p2P, H2, 2048, 512, 512, 1);

    // stage 5: final 1-wide projection -> (T,B,1) fp32
    head3_kernel<<<256, 256, 0, stream>>>(H2, p3_w, p3_b, outp);
}
